// RNNModel_49649821942228
// MI455X (gfx1250) — matrix operand order repaired and checked by reading, not compiled, not run on hardware
//
#include <hip/hip_runtime.h>

// ---------------------------------------------------------------------------
// RNN scan for MI455X (gfx1250, wave32, WMMA).
//   xw_t = x[:,t,:] @ W_ih + b_h      (computed on the fly, no 1GB xw buffer)
//   h    = tanh(xw_t + h @ W_hh)      (1024 sequential steps, h in LDS)
//   out  = h @ W_out + b_out
// Each workgroup owns 16 batch rows for the whole scan (batch-separable
// recurrence -> zero inter-WG sync). 16 waves/WG; wave w owns hidden columns
// [w*32, w*32+32). All GEMMs use v_wmma_f32_16x16x32_bf16 (fp32 accumulate).
// W_hh B-fragments are software-pipelined from L2 (512 KB bf16, L2-resident).
// tanh uses the gfx1250 hardware V_TANH_F32 transcendental when available.
// ---------------------------------------------------------------------------

typedef __bf16 bf16_t;
typedef __attribute__((ext_vector_type(16))) __bf16 v16bf;
typedef __attribute__((ext_vector_type(8)))  __bf16 v8bf;
typedef __attribute__((ext_vector_type(8)))  float  v8f;

namespace {
constexpr int kBatch = 512;
constexpr int kSeq   = 1024;
constexpr int kIn    = 64;
constexpr int kHid   = 512;
constexpr int kOut   = 64;

constexpr int kKtH = kHid / 32;   // 16 k-tiles over HID
constexpr int kNtH = kHid / 16;   // 32 n-tiles over HID
constexpr int kKtI = kIn  / 32;   // 2  k-tiles over IN
constexpr int kNtO = kOut / 16;   // 4  n-tiles over OUT

// Workspace layout (bf16 elements): pre-swizzled WMMA B-fragments.
// Fragment order: [ntile][ktile][lane][16 contiguous K-elements]
constexpr int kWhhOff    = 0;
constexpr int kWhhElems  = kNtH * kKtH * 32 * 16;   // 262144 (512 KB)
constexpr int kWihOff    = kWhhOff + kWhhElems;
constexpr int kWihElems  = kNtH * kKtI * 32 * 16;   // 32768
constexpr int kWoutOff   = kWihOff + kWihElems;
constexpr int kWoutElems = kNtO * kKtH * 32 * 16;   // 32768
constexpr int kWsElems   = kWoutOff + kWoutElems;   // 327680 (~640 KB)

constexpr int kHRow = 520;          // padded LDS row stride (bf16) -> bank spread
constexpr int kHBuf = 16 * kHRow;   // one h buffer (16 rows)
constexpr int kThreads = 512;       // 16 waves
}  // namespace

// ---------------------------------------------------------------------------
// Prep: fp32 row-major weights -> bf16 WMMA B-fragment layout.
// B-fragment (32K x 16N tile): lane L holds column N = nt*16 + L%16,
// K = kt*32 + (L/16)*8 + (e%8) + (e/8)*16 for e in [0,16).
// ---------------------------------------------------------------------------
__global__ void rnn_prep_kernel(const float* __restrict__ Wih,
                                const float* __restrict__ Whh,
                                const float* __restrict__ Wout,
                                bf16_t* __restrict__ ws) {
  int idx = blockIdx.x * blockDim.x + threadIdx.x;
  const float* src;
  bf16_t* dst;
  int f, KT, ncols;
  if (idx < kWhhElems)     { src = Whh;  dst = ws + kWhhOff;  f = idx;            KT = kKtH; ncols = kHid; }
  else if (idx < kWoutOff) { src = Wih;  dst = ws + kWihOff;  f = idx - kWihOff;  KT = kKtI; ncols = kHid; }
  else if (idx < kWsElems) { src = Wout; dst = ws + kWoutOff; f = idx - kWoutOff; KT = kKtH; ncols = kOut; }
  else return;
  const int e    = f & 15;
  const int lane = (f >> 4) & 31;
  const int tile = f >> 9;
  const int kt   = tile % KT;
  const int nt   = tile / KT;
  const int k = kt * 32 + (lane >> 4) * 8 + (e & 7) + (e >> 3) * 16;
  const int n = nt * 16 + (lane & 15);
  dst[f] = (bf16_t)src[(size_t)k * ncols + n];
}

__device__ __forceinline__ v8f splat8(float v) {
  v8f r;
#pragma unroll
  for (int i = 0; i < 8; ++i) r[i] = v;
  return r;
}

// gfx1250 hardware transcendental tanh (TRANS32 op, co-executes with WMMA).
__device__ __forceinline__ float fast_tanh(float v) {
#if __has_builtin(__builtin_amdgcn_tanhf)
  return __builtin_amdgcn_tanhf(v);
#else
  return tanhf(v);
#endif
}

// 16-bit A fragment (16M x 32K): lane L holds row M = L%16,
// K = kb + [0..8) and K = kb + 16 + [0..8) with kb = kt*32 + (L/16)*8.
__device__ __forceinline__ v16bf make_a_lds(const bf16_t* __restrict__ row, int kb) {
  v8bf lo = *(const v8bf*)(row + kb);
  v8bf hi = *(const v8bf*)(row + kb + 16);
  return __builtin_shufflevector(lo, hi, 0, 1, 2, 3, 4, 5, 6, 7,
                                 8, 9, 10, 11, 12, 13, 14, 15);
}

__device__ __forceinline__ v8f wmma_bf16(v16bf a, v16bf b, v8f c) {
  return __builtin_amdgcn_wmma_f32_16x16x32_bf16(false, a, false, b, (short)0, c,
                                                 false, false);
}

// ---------------------------------------------------------------------------
// Main scan kernel: 512 threads = 16 waves, each wave owns 2 n-tiles (32 cols).
// ---------------------------------------------------------------------------
__global__ __launch_bounds__(kThreads) void rnn_scan_kernel(
    const float* __restrict__ x, const float* __restrict__ b_h,
    const float* __restrict__ b_out, const bf16_t* __restrict__ wfrag,
    float* __restrict__ out) {
  __shared__ bf16_t hbuf[2 * kHBuf];

  const int tid  = threadIdx.x;
  const int wave = tid >> 5;          // 0..15
  const int lane = tid & 31;
  const int half = lane >> 4;
  const int l16  = lane & 15;
  const int m0   = blockIdx.x * 16;   // first batch row of this WG
  const int nt0  = wave * 2;          // first 16-col n-tile of this wave

  // h0 = 0
  for (int i = tid; i < 2 * kHBuf; i += kThreads) hbuf[i] = (bf16_t)0.0f;
  __syncthreads();

  const float bh0 = b_h[(nt0 + 0) * 16 + l16];
  const float bh1 = b_h[(nt0 + 1) * 16 + l16];

  // Per-lane fragment base pointers (lane-dependent part folded in once).
  const bf16_t* __restrict__ whhp =
      wfrag + kWhhOff + (size_t)nt0 * kKtH * 512 + (size_t)lane * 16;
  const bf16_t* __restrict__ wihp =
      wfrag + kWihOff + (size_t)nt0 * kKtI * 512 + (size_t)lane * 16;
  const float* __restrict__ xrow = x + (size_t)(m0 + l16) * kSeq * kIn;

  for (int t = 0; t < kSeq; ++t) {
    const bf16_t* __restrict__ hcur = hbuf + (t & 1) * kHBuf;
    bf16_t* __restrict__ hnxt = hbuf + ((t & 1) ^ 1) * kHBuf;
    const float* __restrict__ xt = xrow + (size_t)t * kIn;

    // --- Issue all independent global loads for this step up front ---
    v8f xlo0 = *(const v8f*)(xt + half * 8);            // x k-tile 0
    v8f xhi0 = *(const v8f*)(xt + 16 + half * 8);
    v8f xlo1 = *(const v8f*)(xt + 32 + half * 8);       // x k-tile 1
    v8f xhi1 = *(const v8f*)(xt + 48 + half * 8);
    v16bf bi00 = *(const v16bf*)(wihp + 0 * kKtI * 512 + 0 * 512);  // Wih(nt0,kt0)
    v16bf bi01 = *(const v16bf*)(wihp + 0 * kKtI * 512 + 1 * 512);  // Wih(nt0,kt1)
    v16bf bi10 = *(const v16bf*)(wihp + 1 * kKtI * 512 + 0 * 512);  // Wih(nt1,kt0)
    v16bf bi11 = *(const v16bf*)(wihp + 1 * kKtI * 512 + 1 * 512);  // Wih(nt1,kt1)
    v16bf b0 = *(const v16bf*)(whhp + 0 * kKtH * 512);  // Whh(nt0, kt=0)
    v16bf b1 = *(const v16bf*)(whhp + 1 * kKtH * 512);  // Whh(nt1, kt=0)

    v8f acc0 = splat8(bh0);
    v8f acc1 = splat8(bh1);

    // --- xw_t = x[:,t,:] @ W_ih (K = 64) ---
    v16bf a0, a1;
#pragma unroll
    for (int e = 0; e < 8; ++e) {
      a0[e] = (bf16_t)xlo0[e]; a0[e + 8] = (bf16_t)xhi0[e];
      a1[e] = (bf16_t)xlo1[e]; a1[e + 8] = (bf16_t)xhi1[e];
    }
    acc0 = wmma_bf16(a0, bi00, acc0);
    acc1 = wmma_bf16(a0, bi10, acc1);
    acc0 = wmma_bf16(a1, bi01, acc0);
    acc1 = wmma_bf16(a1, bi11, acc1);

    // --- h @ W_hh (K = 512), software-pipelined B-fragment stream ---
    const bf16_t* __restrict__ ha = hcur + l16 * kHRow;
#pragma unroll 5
    for (int kt = 0; kt < kKtH - 1; ++kt) {
      v16bf n0 = *(const v16bf*)(whhp + 0 * kKtH * 512 + (size_t)(kt + 1) * 512);
      v16bf n1 = *(const v16bf*)(whhp + 1 * kKtH * 512 + (size_t)(kt + 1) * 512);
      v16bf a = make_a_lds(ha, kt * 32 + half * 8);
      acc0 = wmma_bf16(a, b0, acc0);
      acc1 = wmma_bf16(a, b1, acc1);
      b0 = n0;
      b1 = n1;
    }
    {
      v16bf a = make_a_lds(ha, (kKtH - 1) * 32 + half * 8);
      acc0 = wmma_bf16(a, b0, acc0);
      acc1 = wmma_bf16(a, b1, acc1);
    }

    // --- h_next = tanh(acc) -> other LDS buffer ---
    // C/D layout: VGPR r holds row M = half*8 + r, column = ntile*16 + l16.
    bf16_t* __restrict__ hw0 = hnxt + (nt0 + 0) * 16 + l16;
    bf16_t* __restrict__ hw1 = hnxt + (nt0 + 1) * 16 + l16;
#pragma unroll
    for (int r = 0; r < 8; ++r) {
      hw0[(half * 8 + r) * kHRow] = (bf16_t)fast_tanh(acc0[r]);
      hw1[(half * 8 + r) * kHRow] = (bf16_t)fast_tanh(acc1[r]);
    }
    __syncthreads();
  }

  // --- out = h_final @ W_out + b_out (h_final is in buffer 0) ---
  if (wave < kNtO) {
    v8f acc = splat8(b_out[wave * 16 + l16]);
    const bf16_t* __restrict__ woutp =
        wfrag + kWoutOff + (size_t)wave * kKtH * 512 + (size_t)lane * 16;
    const bf16_t* __restrict__ ha = hbuf + l16 * kHRow;
#pragma unroll 4
    for (int kt = 0; kt < kKtH; ++kt) {
      v16bf a = make_a_lds(ha, kt * 32 + half * 8);
      v16bf b = *(const v16bf*)(woutp + (size_t)kt * 512);
      acc = wmma_bf16(a, b, acc);
    }
#pragma unroll
    for (int r = 0; r < 8; ++r) {
      out[(size_t)(m0 + half * 8 + r) * kOut + wave * 16 + l16] = acc[r];
    }
  }
}

extern "C" void kernel_launch(void* const* d_in, const int* in_sizes, int n_in,
                              void* d_out, int out_size, void* d_ws, size_t ws_size,
                              hipStream_t stream) {
  const float* x     = (const float*)d_in[0];   // [512,1024,64]
  const float* W_ih  = (const float*)d_in[1];   // [64,512]
  const float* W_hh  = (const float*)d_in[2];   // [512,512]
  const float* b_h   = (const float*)d_in[3];   // [512]
  const float* W_out = (const float*)d_in[4];   // [512,64]
  const float* b_out = (const float*)d_in[5];   // [64]
  float* outp = (float*)d_out;                  // [512,64]
  bf16_t* wfrag = (bf16_t*)d_ws;                // ~640 KB swizzled bf16 weights

  rnn_prep_kernel<<<(kWsElems + 255) / 256, 256, 0, stream>>>(W_ih, W_hh, W_out, wfrag);
  rnn_scan_kernel<<<kBatch / 16, kThreads, 0, stream>>>(x, b_h, b_out, wfrag, outp);
}
